// HiGCN_79164837200122
// MI455X (gfx1250) — matrix operand order repaired and checked by reading, not compiled, not run on hardware
//
#include <hip/hip_runtime.h>
#include <hip/hip_bf16.h>

typedef __attribute__((ext_vector_type(16))) __bf16 v16bf;
typedef __attribute__((ext_vector_type(8)))  float  v8f;

#define NNODES 50000
#define HDIM   128
#define NEDGE  800000
#define NGRAPH 512
#define NCLASS 10
#define KSTEPS 10
#define BNEPS  1e-5f

__device__ __forceinline__ unsigned short f2bf(float f) {
  unsigned int u = __float_as_uint(f);
  u += 0x7FFFu + ((u >> 16) & 1u);           // round-to-nearest-even
  return (unsigned short)(u >> 16);
}

// ---------------- elementwise helpers ----------------
__global__ void k_convert(const float* __restrict__ src,
                          unsigned short* __restrict__ dst, int n) {
  int i = blockIdx.x * blockDim.x + threadIdx.x;
  if (i < n) dst[i] = f2bf(src[i]);
}

__global__ void k_zero4(float4* __restrict__ p, int n4) {
  int i = blockIdx.x * blockDim.x + threadIdx.x;
  if (i < n4) p[i] = make_float4(0.f, 0.f, 0.f, 0.f);
}

__global__ void k_scale4(float4* __restrict__ out, const float4* __restrict__ x,
                         const float* __restrict__ coef, int ci, int n4) {
  int i = blockIdx.x * blockDim.x + threadIdx.x;
  if (i >= n4) return;
  float c = coef[ci];
  float4 v = x[i];
  out[i] = make_float4(c * v.x, c * v.y, c * v.z, c * v.w);
}

// out += coef[ci]*x ; optionally also emit bf16 copy of the result
__global__ void k_axpy4c(float4* __restrict__ out, const float4* __restrict__ x,
                         const float* __restrict__ coef, int ci, int n4,
                         ushort4* __restrict__ obf) {
  int i = blockIdx.x * blockDim.x + threadIdx.x;
  if (i >= n4) return;
  float c = coef[ci];
  float4 v = x[i];
  float4 o = out[i];
  o.x += c * v.x; o.y += c * v.y; o.z += c * v.z; o.w += c * v.w;
  out[i] = o;
  if (obf) obf[i] = make_ushort4(f2bf(o.x), f2bf(o.y), f2bf(o.z), f2bf(o.w));
}

// ---------------- weight packing into WMMA B-fragment order ----------------
// Wp[((kt*2+g)*128 + n)*16 + e] = bf16( W[(kt*32 + (e<8 ? g*8+e : 16+g*8+e-8))*128 + n] )
__global__ void k_pack_w(const float* __restrict__ W,
                         unsigned short* __restrict__ Wp, int K) {
  int idx = blockIdx.x * blockDim.x + threadIdx.x;
  if (idx >= K * 128) return;
  int e  = idx & 15;
  int t  = idx >> 4;
  int n  = t & 127;
  int kg = t >> 7;          // kt*2 + g
  int g  = kg & 1;
  int kt = kg >> 1;
  int k  = kt * 32 + ((e < 8) ? (g * 8 + e) : (16 + g * 8 + (e - 8)));
  Wp[idx] = f2bf(W[(size_t)k * 128 + n]);
}

// ---------------- WMMA GEMM: C[rows,128] = act(bn(A @ W + bias)) ----------------
// A bf16 [rows,128] (A1 supplies K=128..255 when ktiles==8). A tile staged in LDS.
// Optionally also writes a bf16 copy of C to Cbf.
__global__ __launch_bounds__(128)
void k_gemm(const unsigned short* __restrict__ A0,
            const unsigned short* __restrict__ A1,
            const unsigned short* __restrict__ Wp,
            const float* __restrict__ bias,
            const float* __restrict__ gmm,
            const float* __restrict__ bta,
            const float* __restrict__ mu,
            const float* __restrict__ var,
            float* __restrict__ C,
            unsigned short* __restrict__ Cbf,
            int ktiles, int use_bn, int use_relu) {
  const int lane = threadIdx.x & 31;
  const int wave = threadIdx.x >> 5;
  const int g    = lane >> 4;       // lane group
  const int mr   = lane & 15;
  const int m0   = blockIdx.x * 16;
  const int c0   = wave * 32;
  const int K    = ktiles * 32;     // 128 or 256

  // ---- stage the 16xK bf16 A tile in LDS (shared by all 4 waves) ----
  __shared__ unsigned short Asm[16 * 256];
  const int kchunks = K / 8;                      // uint4 chunks per row
  for (int i = threadIdx.x; i < 16 * kchunks; i += 128) {
    int r  = i / kchunks;
    int k0 = (i % kchunks) * 8;
    const unsigned short* src = (k0 < 128)
        ? A0 + (size_t)(m0 + r) * 128 + k0
        : A1 + (size_t)(m0 + r) * 128 + (k0 - 128);
    *reinterpret_cast<uint4*>(Asm + r * K + k0) =
        *reinterpret_cast<const uint4*>(src);
  }
  __syncthreads();

  v8f acc0 = {};
  v8f acc1 = {};
  union Frag { uint4 u[2]; v16bf v; };
  Frag a, b0, b1;

  for (int kt = 0; kt < ktiles; ++kt) {
    // A fragment from LDS: lane(row=mr, group g) holds
    // K = kt*32 + {g*8..g*8+7, 16+g*8..16+g*8+7}
    const uint4* pa = reinterpret_cast<const uint4*>(Asm + mr * K + kt * 32 + g * 8);
    a.u[0] = pa[0];
    a.u[1] = pa[2];
    // B fragments: packed contiguous 16 bf16 per (kt,g,col)
    const uint4* pb0 = reinterpret_cast<const uint4*>(
        Wp + (((size_t)(kt * 2 + g)) * 128 + (size_t)(c0 + mr)) * 16);
    b0.u[0] = pb0[0]; b0.u[1] = pb0[1];
    const uint4* pb1 = reinterpret_cast<const uint4*>(
        Wp + (((size_t)(kt * 2 + g)) * 128 + (size_t)(c0 + 16 + mr)) * 16);
    b1.u[0] = pb1[0]; b1.u[1] = pb1[1];

    acc0 = __builtin_amdgcn_wmma_f32_16x16x32_bf16(false, a.v, false, b0.v,
                                                   (short)0, acc0, false, false);
    acc1 = __builtin_amdgcn_wmma_f32_16x16x32_bf16(false, a.v, false, b1.v,
                                                   (short)0, acc1, false, false);
  }

  const int col0 = c0 + mr, col1 = c0 + 16 + mr;
  float bs0 = bias[col0], bs1 = bias[col1];
  float sc0 = 1.f, sh0 = 0.f, sc1 = 1.f, sh1 = 0.f;
  if (use_bn) {
    sc0 = gmm[col0] * rsqrtf(var[col0] + BNEPS);
    sh0 = bta[col0] - mu[col0] * sc0;
    sc1 = gmm[col1] * rsqrtf(var[col1] + BNEPS);
    sh1 = bta[col1] - mu[col1] * sc1;
  }
#pragma unroll
  for (int r = 0; r < 8; ++r) {
    int m = m0 + g * 8 + r;              // C layout: VGPR r -> M = r + 8*g, N = lane&15
    float v0 = (acc0[r] + bs0) * sc0 + sh0;
    float v1 = (acc1[r] + bs1) * sc1 + sh1;
    if (use_relu) { v0 = fmaxf(v0, 0.f); v1 = fmaxf(v1, 0.f); }
    C[(size_t)m * 128 + col0] = v0;
    C[(size_t)m * 128 + col1] = v1;
    if (Cbf) {
      Cbf[(size_t)m * 128 + col0] = f2bf(v0);
      Cbf[(size_t)m * 128 + col1] = f2bf(v1);
    }
  }
}

// ---------------- SpMM: out[row] += ew * h[col], one wave per edge ----------------
__global__ __launch_bounds__(256)
void k_spmm(const int* __restrict__ ei, const float* __restrict__ ew,
            const float* __restrict__ h, float* __restrict__ out, int E) {
  int e = blockIdx.x * 8 + (threadIdx.x >> 5);
  if (e >= E) return;
  int lane = threadIdx.x & 31;
  int row = ei[e];
  int col = ei[E + e];
  float w = ew[e];
  // warm the atomic destination line (global_prefetch_b8)
  __builtin_prefetch(out + (size_t)row * 128 + lane * 4, 1, 0);
  float4 v = reinterpret_cast<const float4*>(h + (size_t)col * 128)[lane];
  float* dst = out + (size_t)row * 128 + lane * 4;
  atomicAdd(dst + 0, w * v.x);
  atomicAdd(dst + 1, w * v.y);
  atomicAdd(dst + 2, w * v.z);
  atomicAdd(dst + 3, w * v.w);
}

// ---------------- graph readout: g[batch[n]] += h[n] ----------------
__global__ __launch_bounds__(256)
void k_readout(const int* __restrict__ batch, const float* __restrict__ h,
               float* __restrict__ gsum, int N) {
  int i = blockIdx.x * blockDim.x + threadIdx.x;   // N*32 threads
  int node = i >> 5;
  if (node >= N) return;
  int f4 = (i & 31) * 4;
  int b = batch[node];
  float4 v = *reinterpret_cast<const float4*>(h + (size_t)node * 128 + f4);
  float* dst = gsum + (size_t)b * 128 + f4;
  atomicAdd(dst + 0, v.x);
  atomicAdd(dst + 1, v.y);
  atomicAdd(dst + 2, v.z);
  atomicAdd(dst + 3, v.w);
}

// ---------------- tiny final linear [512,128]@[128,10] ----------------
__global__ void k_lin2(const float* __restrict__ g2, const float* __restrict__ W,
                       const float* __restrict__ b, float* __restrict__ out) {
  int idx = blockIdx.x * blockDim.x + threadIdx.x;
  if (idx >= NGRAPH * NCLASS) return;
  int r = idx / NCLASS, c = idx % NCLASS;
  float s = b[c];
  const float* gr = g2 + (size_t)r * 128;
#pragma unroll 4
  for (int k = 0; k < 128; ++k) s += gr[k] * W[k * NCLASS + c];
  out[idx] = s;
}

extern "C" void kernel_launch(void* const* d_in, const int* in_sizes, int n_in,
                              void* d_out, int out_size, void* d_ws, size_t ws_size,
                              hipStream_t stream) {
  if (n_in < 70) return;
  (void)in_sizes; (void)out_size;

  const size_t NH  = (size_t)NNODES * HDIM;   // 6,400,000
  const int    NH4 = (int)(NH / 4);           // 1,600,000
  const int    GRD = (int)(NH4 / 256);        // 6250
  const int    CNV = (int)(NH / 256);         // 25000

  // -------- workspace layout --------
  float* F0 = (float*)d_ws;
  float* F1 = F0 + NH;
  float* F2 = F1 + NH;
  float* F3 = F2 + NH;
  unsigned short* B0 = (unsigned short*)(F3 + NH);
  unsigned short* B1 = B0 + NH;
  unsigned short* B2 = B1 + NH;
  unsigned short* B3 = B2 + NH;
  unsigned short* WP = B3 + NH;
  const size_t LAYER_WP = 16384 + 16384 + 32768 + 16384 + 16384; // 98304
  unsigned short* lin1P = WP + 3 * LAYER_WP;
  float* Gsum = (float*)(lin1P + 16384);
  float* G2   = Gsum + (size_t)NGRAPH * HDIM;
  unsigned short* Gbf = (unsigned short*)(G2 + (size_t)NGRAPH * HDIM);
  size_t need = (size_t)((char*)(Gbf + (size_t)NGRAPH * HDIM) - (char*)d_ws);
  if (ws_size < need) return;

  // -------- inputs --------
  const float* x     = (const float*)d_in[0];
  const int*   ei1   = (const int*)d_in[1];
  const float* ew1   = (const float*)d_in[2];
  const int*   ei2   = (const int*)d_in[3];
  const float* ew2   = (const float*)d_in[4];
  const int*   batch = (const int*)d_in[5];
  // per-layer params, sorted dict keys:
  // 0:W1 1:W2 2:Wi0 3:Wi1 4:Wo 5:b1 6:b2 7:be1 8:be2 9:bi0 10:bi1 11:bo
  // 12:fW0 13:fW1 14:g1 15:g2 16:m1 17:m2 18:v1 19:v2
  auto P = [&](int l, int j) { return (const float*)d_in[6 + l * 20 + j]; };
  const float* lin1_W = (const float*)d_in[66];
  const float* lin1_b = (const float*)d_in[67];
  const float* lin2_W = (const float*)d_in[68];
  const float* lin2_b = (const float*)d_in[69];

  // -------- pack all weights to bf16 fragment order --------
  for (int l = 0; l < 3; ++l) {
    unsigned short* base = WP + (size_t)l * LAYER_WP;
    k_pack_w<<<64, 256, 0, stream>>>(P(l, 2), base, 128);              // Wi0
    k_pack_w<<<64, 256, 0, stream>>>(P(l, 3), base + 16384, 128);      // Wi1
    k_pack_w<<<128, 256, 0, stream>>>(P(l, 4), base + 32768, 256);     // Wo
    k_pack_w<<<64, 256, 0, stream>>>(P(l, 0), base + 65536, 128);      // W1
    k_pack_w<<<64, 256, 0, stream>>>(P(l, 1), base + 81920, 128);      // W2
  }
  k_pack_w<<<64, 256, 0, stream>>>(lin1_W, lin1P, 128);

  auto gemm = [&](const unsigned short* A0p, const unsigned short* A1p,
                  const unsigned short* Wp, const float* bias,
                  const float* gm, const float* bt, const float* mu, const float* vr,
                  float* Cout, unsigned short* Cbf, int rows, int ktiles,
                  int ubn, int urelu) {
    k_gemm<<<rows / 16, 128, 0, stream>>>(A0p, A1p, Wp, bias, gm, bt, mu, vr,
                                          Cout, Cbf, ktiles, ubn, urelu);
  };

  auto run_prop = [&](float* xx, float* outb, float* tmp,
                      const int* ei, const float* ew, const float* fw,
                      unsigned short* bfout) {
    k_scale4<<<GRD, 256, 0, stream>>>((float4*)outb, (const float4*)xx, fw, 0, NH4);
    float* ping = xx;
    float* pong = tmp;
    for (int k = 0; k < KSTEPS; ++k) {
      k_zero4<<<GRD, 256, 0, stream>>>((float4*)pong, NH4);
      k_spmm<<<NEDGE / 8, 256, 0, stream>>>(ei, ew, ping, pong, NEDGE);
      k_axpy4c<<<GRD, 256, 0, stream>>>(
          (float4*)outb, (const float4*)pong, fw, k + 1, NH4,
          (k == KSTEPS - 1) ? (ushort4*)bfout : (ushort4*)nullptr);
      float* t = ping; ping = pong; pong = t;
    }
  };

  // -------- layers --------
  for (int l = 0; l < 3; ++l) {
    unsigned short* base = WP + (size_t)l * LAYER_WP;
    if (l == 0) k_convert<<<CNV, 256, 0, stream>>>(x, B0, (int)NH);
    // (for l>0, B0 = bf16 activations written by previous layer's W2 gemm)

    // branch 0 (edges 1): xx = h@Wi0+bi0 ; prop -> F2, bf16 -> B1
    gemm(B0, nullptr, base, P(l, 9), 0, 0, 0, 0, F1, nullptr, NNODES, 4, 0, 0);
    run_prop(F1, F2, F3, ei1, ew1, P(l, 12), B1);

    // branch 1 (edges 2): prop -> F2, bf16 -> B2
    gemm(B0, nullptr, base + 16384, P(l, 10), 0, 0, 0, 0, F1, nullptr, NNODES, 4, 0, 0);
    run_prop(F1, F2, F3, ei2, ew2, P(l, 13), B2);

    // concat -> Wo (K = 256): f32 F1 (unused) + bf16 B3
    gemm(B1, B2, base + 32768, P(l, 11), 0, 0, 0, 0, F1, B3, NNODES, 8, 0, 0);

    // MLP: W1 + BN1 + ReLU -> F2 + bf16 B1 ; W2 + BN2 + ReLU -> F0 + bf16 B0
    gemm(B3, nullptr, base + 65536, P(l, 5), P(l, 14), P(l, 7), P(l, 16), P(l, 18),
         F2, B1, NNODES, 4, 1, 1);
    gemm(B1, nullptr, base + 81920, P(l, 6), P(l, 15), P(l, 8), P(l, 17), P(l, 19),
         F0, B0, NNODES, 4, 1, 1);
  }

  // -------- readout --------
  k_zero4<<<64, 256, 0, stream>>>((float4*)Gsum, NGRAPH * HDIM / 4);
  k_readout<<<(NNODES * 32) / 256, 256, 0, stream>>>(batch, F0, Gsum, NNODES);
  k_convert<<<NGRAPH * HDIM / 256, 256, 0, stream>>>(Gsum, Gbf, NGRAPH * HDIM);
  gemm(Gbf, nullptr, lin1P, lin1_b, 0, 0, 0, 0, G2, nullptr, NGRAPH, 4, 0, 1);
  k_lin2<<<(NGRAPH * NCLASS + 255) / 256, 256, 0, stream>>>(G2, lin2_W, lin2_b,
                                                            (float*)d_out);
}
